// CrossAttentionCamWeighter_61383672594793
// MI455X (gfx1250) — matrix operand order repaired and checked by reading, not compiled, not run on hardware
//
#include <hip/hip_runtime.h>
#include <hip/hip_bf16.h>

typedef __attribute__((ext_vector_type(16))) _Float16 v16h;
typedef __attribute__((ext_vector_type(8)))  float    v8f;

#define SB 1024
#define NS 64
#define BT 256
#define HD 128

// A-operand (16x32 f16) LDS index for element (M,K), K in [0,128).
// Contiguous K order: lane = M + ((K/16)&1)*16,
// half index h = K&15 within the 16-half (32B) per-lane slice.
__device__ __forceinline__ int aIndex(int M, int K) {
  int kc = K >> 5, kl = K & 31;
  return kc * 512 + (M + ((kl >> 4) << 4)) * 16 + (kl & 15);
}

__device__ __forceinline__ v8f wmma_step(v16h a, v16h b, v8f c) {
  return __builtin_amdgcn_wmma_f32_16x16x32_f16(false, a, false, b, (short)0, c,
                                                false, false);
}

// One 16-point tile: features -> 42->64->64->128 MLP, all GEMMs via WMMA.
// mode 0: write xk row-major f16 to out3 (sK tile base)
// mode 1: write xq in A-operand layout into out3 (per-wave scratch, 4 K-chunks)
__device__ __forceinline__ void mlp_tile(const float* __restrict__ poseb,
                                         _Float16* wf,
                                         const _Float16* sW1, const _Float16* sW2,
                                         const _Float16* sW3, const float* sB,
                                         int lane, int mode, _Float16* out3)
{
  // ---- features: 2 lanes per point, 21 features each; pad 42..63 with 0 ----
  {
    const int p = lane & 15, hh = lane >> 4;
    const float* pose = poseb + p * 16;
    float loc[3] = {pose[3], pose[7], pose[11]};
    float dir[3] = {pose[2], pose[6], pose[10]};
    v16h z = {};
    *(v16h*)&wf[lane * 16]       = z;   // zero chunk0 (this lane's slice)
    *(v16h*)&wf[512 + lane * 16] = z;   // zero chunk1
    const int k0 = hh * 21;
#pragma unroll
    for (int i = 0; i < 21; ++i) {
      int K = k0 + i;
      float v;
      if (K < 3) {
        v = loc[K];
      } else if (K < 39) {
        int j = (K - 3) / 3, c = (K - 3) % 3;     // emb[j*3+c]
        float f = 1.5f * (float)(1 << (j >> 1));  // 1.5 * 2^(j/2)
        float x = loc[c] * f;
        v = (j & 1) ? __cosf(x) : __sinf(x);      // odd j: phase pi/2
      } else {
        v = dir[K - 39];
      }
      wf[aIndex(p, K)] = (_Float16)v;
    }
  }

  // ---- layers 1,2: 64-wide, bias folded into C init, ReLU, scatter back ----
#pragma unroll
  for (int layer = 0; layer < 2; ++layer) {
    const _Float16* w = layer ? sW2 : sW1;
    const int bofs = layer ? 64 : 0;
    v16h a0 = *(const v16h*)&wf[lane * 16];
    v16h a1 = *(const v16h*)&wf[512 + lane * 16];
#pragma unroll
    for (int nt = 0; nt < 4; ++nt) {
      float bb = sB[bofs + nt * 16 + (lane & 15)];
      v8f c = {bb, bb, bb, bb, bb, bb, bb, bb};
      v16h w0 = *(const v16h*)&w[nt * 512 + lane * 16];
      v16h w1 = *(const v16h*)&w[(4 + nt) * 512 + lane * 16];
      c = wmma_step(a0, w0, c);
      c = wmma_step(a1, w1, c);
      const int Mb = (lane >> 4) << 3;
      const int N  = nt * 16 + (lane & 15);
#pragma unroll
      for (int r = 0; r < 8; ++r)
        wf[aIndex(Mb + r, N)] = (_Float16)fmaxf(c[r], 0.0f);
    }
  }

  // ---- layer 3: 128-wide output ----
  {
    v16h a0 = *(const v16h*)&wf[lane * 16];
    v16h a1 = *(const v16h*)&wf[512 + lane * 16];
#pragma unroll
    for (int nt = 0; nt < 8; ++nt) {
      float bb = sB[128 + nt * 16 + (lane & 15)];
      v8f c = {bb, bb, bb, bb, bb, bb, bb, bb};
      v16h w0 = *(const v16h*)&sW3[nt * 512 + lane * 16];
      v16h w1 = *(const v16h*)&sW3[(8 + nt) * 512 + lane * 16];
      c = wmma_step(a0, w0, c);
      c = wmma_step(a1, w1, c);
      const int Mb = (lane >> 4) << 3;
      const int N  = nt * 16 + (lane & 15);
#pragma unroll
      for (int r = 0; r < 8; ++r) {
        _Float16 hv = (_Float16)c[r];
        if (mode == 0) out3[(Mb + r) * HD + N] = hv;   // xk row-major
        else           out3[aIndex(Mb + r, N)] = hv;   // xq A-layout
      }
    }
  }
}

__global__ __launch_bounds__(256)
void camweight_kernel(const float* __restrict__ input_poses,
                      const float* __restrict__ target_poses,
                      const float* __restrict__ W1, const float* __restrict__ b1,
                      const float* __restrict__ W2, const float* __restrict__ b2,
                      const float* __restrict__ W3, const float* __restrict__ b3,
                      float* __restrict__ out)
{
  __shared__ __align__(32) _Float16 sW1[4096];      // 2 kc x 4 nt B tiles
  __shared__ __align__(32) _Float16 sW2[4096];
  __shared__ __align__(32) _Float16 sW3[8192];      // 2 kc x 8 nt B tiles
  __shared__ float                  sB[256];        // b1|b2|b3
  __shared__ __align__(32) _Float16 sK[NS * HD];    // xk row-major f16
  __shared__ __align__(32) _Float16 sWv[8][2048];   // per-wave A scratch (aliased)

  const int s    = blockIdx.x;
  const int tid  = threadIdx.x;
  const int wv   = tid >> 5;
  const int lane = tid & 31;

  // ---- stage weights into B-operand lane layout (K padded 42->64 for W1) ----
  {
    const int kc = tid >> 7, nt = (tid >> 5) & 3, ln = tid & 31;
    const int Ng = nt * 16 + (ln & 15);
#pragma unroll
    for (int h = 0; h < 16; ++h) {
      int Kg = kc * 32 + ((ln >> 4) << 4) + h;
      sW1[tid * 16 + h] = (_Float16)((Kg < 42) ? W1[Kg * 64 + Ng] : 0.0f);
      sW2[tid * 16 + h] = (_Float16)W2[Kg * 64 + Ng];
    }
#pragma unroll
    for (int t = 0; t < 2; ++t) {
      int i3 = tid + t * 256;
      int kc3 = i3 >> 8, nt3 = (i3 >> 5) & 7, ln3 = i3 & 31;
      int Ng3 = nt3 * 16 + (ln3 & 15);
#pragma unroll
      for (int h = 0; h < 16; ++h) {
        int Kg = kc3 * 32 + ((ln3 >> 4) << 4) + h;
        sW3[i3 * 16 + h] = (_Float16)W3[Kg * 128 + Ng3];
      }
    }
    sB[tid] = (tid < 64) ? b1[tid] : (tid < 128) ? b2[tid - 64] : b3[tid - 128];
  }
  __syncthreads();

  // ---- phase 1: xk for 4 key tiles (waves 0..3) ----
  if (wv < 4) {
    mlp_tile(input_poses + ((size_t)s * NS + wv * 16) * 16,
             sWv[wv], sW1, sW2, sW3, sB, lane, 0, &sK[wv * 16 * HD]);
  }
  __syncthreads();

  // ---- phase 2: per query tile: MLP -> scores (K=128) -> softmax -> store ----
  const float iscale = 0.088388347648318447f;  // 1/sqrt(128)
  for (int qt = wv; qt < 16; qt += 8) {
    _Float16* wf = sWv[wv];   // first 1024: feats/h scratch; all 2048: xq A-layout
    if (qt + 8 < 16)          // prefetch next tile's poses (global_prefetch_b8)
      __builtin_prefetch(target_poses + ((size_t)s * BT + (qt + 8) * 16) * 16, 0, 0);

    mlp_tile(target_poses + ((size_t)s * BT + qt * 16) * 16,
             wf, sW1, sW2, sW3, sB, lane, 1, wf);

    v16h A0 = *(const v16h*)&wf[0 * 512 + lane * 16];
    v16h A1 = *(const v16h*)&wf[1 * 512 + lane * 16];
    v16h A2 = *(const v16h*)&wf[2 * 512 + lane * 16];
    v16h A3 = *(const v16h*)&wf[3 * 512 + lane * 16];

    v8f cs[4];
#pragma unroll
    for (int kt = 0; kt < 4; ++kt) {
      const int key = kt * 16 + (lane & 15);
      const _Float16* kb = &sK[key * HD + ((lane >> 4) << 4)];  // B: contiguous 32B
      v8f c = {};
      c = wmma_step(A0, *(const v16h*)(kb +  0), c);
      c = wmma_step(A1, *(const v16h*)(kb + 32), c);
      c = wmma_step(A2, *(const v16h*)(kb + 64), c);
      c = wmma_step(A3, *(const v16h*)(kb + 96), c);
#pragma unroll
      for (int r = 0; r < 8; ++r) c[r] *= iscale;
      cs[kt] = c;
    }

    // softmax over 64 keys: 4 tiles in regs x 16 lanes of the half-wave
#pragma unroll
    for (int r = 0; r < 8; ++r) {
      float m = fmaxf(fmaxf(cs[0][r], cs[1][r]), fmaxf(cs[2][r], cs[3][r]));
      m = fmaxf(m, __shfl_xor(m, 1, 16));
      m = fmaxf(m, __shfl_xor(m, 2, 16));
      m = fmaxf(m, __shfl_xor(m, 4, 16));
      m = fmaxf(m, __shfl_xor(m, 8, 16));
      float sum = 0.0f;
#pragma unroll
      for (int kt = 0; kt < 4; ++kt) {
        float e = __expf(cs[kt][r] - m);
        cs[kt][r] = e;
        sum += e;
      }
      sum += __shfl_xor(sum, 1, 16);
      sum += __shfl_xor(sum, 2, 16);
      sum += __shfl_xor(sum, 4, 16);
      sum += __shfl_xor(sum, 8, 16);
      float inv = 1.0f / sum;
#pragma unroll
      for (int kt = 0; kt < 4; ++kt) cs[kt][r] *= inv;
    }

    // transposed store out[s][key][q]: 8 consecutive q per lane -> 2x float4
    float* ob = out + (size_t)s * NS * BT + qt * 16 + ((lane >> 4) << 3);
#pragma unroll
    for (int kt = 0; kt < 4; ++kt) {
      const int key = kt * 16 + (lane & 15);
      float4 lo = make_float4(cs[kt][0], cs[kt][1], cs[kt][2], cs[kt][3]);
      float4 hi = make_float4(cs[kt][4], cs[kt][5], cs[kt][6], cs[kt][7]);
      *(float4*)&ob[key * BT]     = lo;
      *(float4*)&ob[key * BT + 4] = hi;
    }
  }
}

extern "C" void kernel_launch(void* const* d_in, const int* in_sizes, int n_in,
                              void* d_out, int out_size, void* d_ws, size_t ws_size,
                              hipStream_t stream) {
  const float* input_poses  = (const float*)d_in[0];
  const float* target_poses = (const float*)d_in[1];
  const float* W1 = (const float*)d_in[2];
  const float* b1 = (const float*)d_in[3];
  const float* W2 = (const float*)d_in[4];
  const float* b2 = (const float*)d_in[5];
  const float* W3 = (const float*)d_in[6];
  const float* b3 = (const float*)d_in[7];
  float* out = (float*)d_out;

  hipLaunchKernelGGL(camweight_kernel, dim3(SB), dim3(256), 0, stream,
                     input_poses, target_poses, W1, b1, W2, b2, W3, b3, out);
}
